// DKM_12936441495853
// MI455X (gfx1250) — compile-verified, matrix operand order checked by reading, not run on hardware
//
#include <hip/hip_runtime.h>

#define NB_N 65536
#define NB_D 64
#define NB_K 256
#define DKM_ITERS 10
#define DKM_EPS 1e-4f

typedef __attribute__((ext_vector_type(16))) __bf16 v16bf;
typedef __attribute__((ext_vector_type(8)))  float  v8f;
typedef __attribute__((ext_vector_type(8)))  unsigned short ush8;
typedef __attribute__((ext_vector_type(4)))  int    v4i;
typedef __attribute__((ext_vector_type(8)))  int    v8i;

union FragU { ush8 h[2]; v16bf v; };

// f32 -> bf16 round-to-nearest-even
__device__ __forceinline__ unsigned short f2bf(float x) {
  union { float f; unsigned u; } v; v.f = x;
  unsigned r = v.u + 0x7FFFu + ((v.u >> 16) & 1u);
  return (unsigned short)(r >> 16);
}

// Load a 16-half WMMA operand fragment from LDS.
// Per ISA layout: lanes 0-15 hold K 0..7 and 16..23, lanes 16-31 hold K 8..15
// and 24..31 within the 32-wide inner chunk -> two b128 LDS reads.
__device__ __forceinline__ v16bf ldfrag(const unsigned short* rowptr, int hs) {
  FragU f;
  const ush8* p = (const ush8*)(rowptr + hs * 8);
  f.h[0] = p[0];
  f.h[1] = p[2];
  return f.v;
}

__device__ __forceinline__ v8f wmma_bf16(v16bf a, v16bf b, v8f c) {
  return __builtin_amdgcn_wmma_f32_16x16x32_bf16(false, a, false, b, (short)0, c,
                                                 false, false);
}

// Issue a TDM DMA of `bytes/4` contiguous DWORDs from global `src` into LDS at
// `ldsoff`, as a 1-row 2D tile. D# layout per CDNA5 ISA 8.3/8.4.
// Must be executed by a single wave; EXEC is ignored by TENSOR ops.
__device__ __forceinline__ void tdm_load_1d(const void* src, unsigned ldsoff,
                                            unsigned nelem_dw) {
  unsigned long long ga = (unsigned long long)(uintptr_t)src;
  v4i g0;
  g0.x = 1;                                            // count=1, user D#
  g0.y = (int)ldsoff;                                  // lds_addr (bytes)
  g0.z = (int)(unsigned)(ga & 0xffffffffu);            // global_addr[31:0]
  g0.w = (int)(((ga >> 32) & 0x01ffffffu) | (2u << 30)); // addr[56:32] | type=2
  v8i g1;
  g1[0] = 0x20000;                  // wg_mask=0 (not in cluster), data_size=2 (4B)
  g1[1] = (int)(nelem_dw << 16);    // tensor_dim0[15:0] in bits 63:48
  g1[2] = (int)((nelem_dw >> 16) & 0xffffu) | 0x10000; // dim0[31:16], tensor_dim1=1
  g1[3] = (int)(nelem_dw << 16);    // tile_dim0 in bits 127:112 (fits 16b)
  g1[4] = 1;                        // tile_dim1=1, tile_dim2=0
  g1[5] = (int)nelem_dw;            // tensor_dim0_stride[31:0]
  g1[6] = 0;                        // stride0[47:32]=0, stride1 lo=0
  g1[7] = 0;
  asm volatile("tensor_load_to_lds %0, %1" :: "s"(g0), "s"(g1) : "memory");
  __builtin_amdgcn_s_wait_tensorcnt(0);
}

// ---------------------------------------------------------------------------
// k_init: clusters0 = W[init_idx], AC = clusters0, done = 0
// grid: 256 blocks x 64 threads
// ---------------------------------------------------------------------------
__global__ void k_init(const float* __restrict__ W, const int* __restrict__ idx,
                       float* __restrict__ C, float* __restrict__ AC,
                       int* __restrict__ done) {
  int k = blockIdx.x, d = threadIdx.x;
  int row = idx[k];
  float v = W[(size_t)row * NB_D + d];
  C[k * NB_D + d]  = v;
  AC[k * NB_D + d] = v;
  if (k == 0 && d == 0) *done = 0;
}

// grid: 64 x 256  (zeros wsum[16384] and csum[256])
__global__ void k_zero(float* __restrict__ wsum, float* __restrict__ csum) {
  int i = blockIdx.x * blockDim.x + threadIdx.x;
  if (i < NB_K * NB_D) wsum[i] = 0.f;
  if (i < NB_K) csum[i] = 0.f;
}

// ---------------------------------------------------------------------------
// k_attn_iter: attn = softmax(-dist(W, C)); csum += col-sums of attn
// grid: N/128 blocks x 128 threads (4 waves; each wave does two 16-row tiles)
// dyn LDS: stage[256*64]f32 | cl[256*64]bf16 | wt[4*16*64]bf16
//          | c2[256]f32 | w2s[64]f32 | csl[256]f32
// Cluster table arrives via the Tensor Data Mover (one DMA per block).
// ---------------------------------------------------------------------------
__global__ void __launch_bounds__(128, 1)
k_attn_iter(const float* __restrict__ W, const float* __restrict__ C,
            float* __restrict__ attn_out, float* __restrict__ csum,
            const int* __restrict__ done) {
  if (*done) return;
  extern __shared__ char smem[];
  float* stage = (float*)smem;                                // 256*64 f32
  unsigned short* cl = (unsigned short*)(stage + NB_K * NB_D); // 256*64 bf16
  unsigned short* wt = cl + NB_K * NB_D;                      // 4*16*64
  float* c2  = (float*)(wt + 4 * 16 * NB_D);                  // 256
  float* w2s = c2 + NB_K;                                     // 64
  float* csl = w2s + 64;                                      // 256

  int tid = threadIdx.x;
  int lane = tid & 31, wave = tid >> 5;
  int ln = lane & 15, hs = lane >> 4;

  // TDM: DMA the f32 cluster table (64KB) into LDS; wave 0 issues + waits.
  if (tid < 32)
    tdm_load_1d(C, (unsigned)(uintptr_t)(void*)stage, NB_K * NB_D);
  __syncthreads();

  // Convert staged f32 -> bf16 table, and compute ||c||^2 per cluster from LDS.
  for (int i = tid; i < NB_K * NB_D; i += blockDim.x) cl[i] = f2bf(stage[i]);
  for (int k = tid; k < NB_K; k += blockDim.x) {
    float s = 0.f;
    for (int d = 0; d < NB_D; ++d) { float v = stage[k * NB_D + d]; s += v * v; }
    c2[k] = s;
  }
  for (int i = tid; i < NB_K; i += blockDim.x) csl[i] = 0.f;
  __syncthreads();

  unsigned short* wtw = wt + wave * 16 * NB_D;
  float* w2w = w2s + wave * 16;

  for (int tl = 0; tl < 2; ++tl) {
    int rowbase = blockIdx.x * 128 + tl * 64 + wave * 16;

    // zero per-wave ||w||^2 slots (same-wave LDS ordering is guaranteed)
    if (lane < 16) w2w[lane] = 0.f;
    // Load + convert this wave's 16x64 W tile; accumulate ||w||^2 per row.
    for (int i = 0; i < 32; ++i) {
      int idx = i * 32 + lane;
      float v = W[(size_t)rowbase * NB_D + idx];
      wtw[idx] = f2bf(v);
      atomicAdd(&w2w[idx >> 6], v * v);
    }
    __builtin_prefetch(W + (size_t)((rowbase + 64) % NB_N) * NB_D, 0, 3);

    v8f zf = {0.f, 0.f, 0.f, 0.f, 0.f, 0.f, 0.f, 0.f};
    v8f acc[16];
#pragma unroll
    for (int t = 0; t < 16; ++t) acc[t] = zf;

    // GEMM1: S(16x256) = Wtile(16x64) @ C^T, inner dim in 2 chunks of 32.
#pragma unroll
    for (int c = 0; c < 2; ++c) {
      v16bf a = ldfrag(wtw + ln * NB_D + c * 32, hs);
#pragma unroll
      for (int t = 0; t < 16; ++t) {
        v16bf b = ldfrag(cl + (t * 16 + ln) * NB_D + c * 32, hs);
        acc[t] = wmma_bf16(a, b, acc[t]);
      }
    }

    // scores: -sqrt(max(w2 - 2*dot + c2, 0)); TEMPERATURE = 1
    float w2v[8];
#pragma unroll
    for (int r = 0; r < 8; ++r) w2v[r] = w2w[r + 8 * hs];
#pragma unroll
    for (int t = 0; t < 16; ++t) {
      float c2t = c2[t * 16 + ln];
#pragma unroll
      for (int r = 0; r < 8; ++r) {
        float d2 = fmaxf(w2v[r] + c2t - 2.0f * acc[t][r], 0.f);
        acc[t][r] = -sqrtf(d2);
      }
    }

    // softmax across K=256 (rows live in fixed VGPR idx across 16 lanes x 16 tiles)
    float mx[8], sm[8];
#pragma unroll
    for (int r = 0; r < 8; ++r) mx[r] = -3.402823466e38f;
#pragma unroll
    for (int t = 0; t < 16; ++t)
#pragma unroll
      for (int r = 0; r < 8; ++r) mx[r] = fmaxf(mx[r], acc[t][r]);
#pragma unroll
    for (int off = 1; off < 16; off <<= 1)
#pragma unroll
      for (int r = 0; r < 8; ++r) mx[r] = fmaxf(mx[r], __shfl_xor(mx[r], off, 16));
#pragma unroll
    for (int r = 0; r < 8; ++r) sm[r] = 0.f;
#pragma unroll
    for (int t = 0; t < 16; ++t)
#pragma unroll
      for (int r = 0; r < 8; ++r) {
        float e = __expf(acc[t][r] - mx[r]);
        acc[t][r] = e;
        sm[r] += e;
      }
#pragma unroll
    for (int off = 1; off < 16; off <<= 1)
#pragma unroll
      for (int r = 0; r < 8; ++r) sm[r] += __shfl_xor(sm[r], off, 16);
#pragma unroll
    for (int r = 0; r < 8; ++r) sm[r] = 1.0f / sm[r];

    // write attn, accumulate csum partials in LDS
#pragma unroll
    for (int t = 0; t < 16; ++t) {
      float col = 0.f;
#pragma unroll
      for (int r = 0; r < 8; ++r) {
        float a = acc[t][r] * sm[r];
        attn_out[(size_t)(rowbase + r + 8 * hs) * NB_K + t * 16 + ln] = a;
        col += a;
      }
      atomicAdd(&csl[t * 16 + ln], col);
    }
  }
  __syncthreads();
  for (int i = tid; i < NB_K; i += blockDim.x) atomicAdd(&csum[i], csl[i]);
}

// ---------------------------------------------------------------------------
// k_wsum_iter: wsum(256x64) += attn^T @ W over a chunk of 1024 rows.
// grid: 64 x 128. dyn LDS: attn_t[256][32]bf16 | Wt[64][32]bf16
// ---------------------------------------------------------------------------
__global__ void __launch_bounds__(128, 1)
k_wsum_iter(const float* __restrict__ attn, const float* __restrict__ W,
            float* __restrict__ wsum, const int* __restrict__ done) {
  if (*done) return;
  extern __shared__ char smem[];
  unsigned short* at = (unsigned short*)smem;   // [256][32] attn^T
  unsigned short* bw = at + NB_K * 32;          // [64][32]  W^T

  int tid = threadIdx.x, lane = tid & 31, wave = tid >> 5;
  int ln = lane & 15, hs = lane >> 4;

  v8f zf = {0.f, 0.f, 0.f, 0.f, 0.f, 0.f, 0.f, 0.f};
  v8f acc[4][4];
#pragma unroll
  for (int kt = 0; kt < 4; ++kt)
#pragma unroll
    for (int dt = 0; dt < 4; ++dt) acc[kt][dt] = zf;

  int base = blockIdx.x * 1024;
  for (int it = 0; it < 32; ++it) {
    int nb = base + it * 32;
    __syncthreads();
    for (int i = tid; i < NB_K * 32; i += blockDim.x) {
      int r = i >> 8, k = i & 255;
      at[k * 32 + r] = f2bf(attn[(size_t)(nb + r) * NB_K + k]);
    }
    for (int i = tid; i < NB_D * 32; i += blockDim.x) {
      int r = i >> 6, d = i & 63;
      bw[d * 32 + r] = f2bf(W[(size_t)(nb + r) * NB_D + d]);
    }
    __syncthreads();
#pragma unroll
    for (int kt = 0; kt < 4; ++kt) {
      v16bf a = ldfrag(at + ((wave * 4 + kt) * 16 + ln) * 32, hs);
#pragma unroll
      for (int dt = 0; dt < 4; ++dt) {
        v16bf b = ldfrag(bw + (dt * 16 + ln) * 32, hs);
        acc[kt][dt] = wmma_bf16(a, b, acc[kt][dt]);
      }
    }
  }
#pragma unroll
  for (int kt = 0; kt < 4; ++kt)
#pragma unroll
    for (int dt = 0; dt < 4; ++dt)
#pragma unroll
      for (int r = 0; r < 8; ++r) {
        int kc = (wave * 4 + kt) * 16 + r + 8 * hs;
        int dc = dt * 16 + ln;
        atomicAdd(&wsum[kc * NB_D + dc], acc[kt][dt][r]);
      }
}

// ---------------------------------------------------------------------------
// k_update: new = wsum/csum; conv = ||C-new||_F <= eps; AC=C; C = conv?C:new
// grid: 1 x 256
// ---------------------------------------------------------------------------
__global__ void k_update(float* __restrict__ C, float* __restrict__ AC,
                         float* __restrict__ wsum, const float* __restrict__ csum,
                         int* __restrict__ done) {
  if (*done) return;
  __shared__ float red[256];
  __shared__ int convs;
  int tid = threadIdx.x;
  float local = 0.f;
  for (int i = tid; i < NB_K * NB_D; i += 256) {
    float nw = wsum[i] / csum[i >> 6];
    float d = C[i] - nw;
    local += d * d;
    wsum[i] = nw;
  }
  red[tid] = local;
  __syncthreads();
  for (int s = 128; s > 0; s >>= 1) {
    if (tid < s) red[tid] += red[tid + s];
    __syncthreads();
  }
  if (tid == 0) {
    int cv = (sqrtf(red[0]) <= DKM_EPS) ? 1 : 0;
    convs = cv;
    if (cv) *done = 1;
  }
  __syncthreads();
  int cv = convs;
  for (int i = tid; i < NB_K * NB_D; i += 256) {
    float c = C[i];
    AC[i] = c;                 // clusters used for last attn (pre-update)
    if (!cv) C[i] = wsum[i];
  }
}

// cn = L2-normalized rows of C. grid: 1 x 256
__global__ void k_norm(const float* __restrict__ C, float* __restrict__ cn) {
  int k = threadIdx.x;
  float s = 0.f;
  for (int d = 0; d < NB_D; ++d) { float v = C[k * NB_D + d]; s += v * v; }
  float inv = 1.0f / fmaxf(sqrtf(s), 1e-12f);
  for (int d = 0; d < NB_D; ++d) cn[k * NB_D + d] = C[k * NB_D + d] * inv;
}

// ---------------------------------------------------------------------------
// k_final: attn = softmax(-dist(W, AC)) (written out) and compressed = attn @ cn
// grid: N/128 x 128 (each wave does two 16-row tiles). dyn LDS:
//   cl[256*64] | cnT[64*256] | wt[4*16*64] | at[4*16*256] (bf16) | c2[256] | w2s[64]
// ---------------------------------------------------------------------------
__global__ void __launch_bounds__(128, 1)
k_final(const float* __restrict__ W, const float* __restrict__ AC,
        const float* __restrict__ cn, float* __restrict__ attn_out,
        float* __restrict__ comp) {
  extern __shared__ char smem[];
  unsigned short* cl  = (unsigned short*)smem;       // 256*64
  unsigned short* cnT = cl + NB_K * NB_D;            // 64*256 (cnT[d][k])
  unsigned short* wt  = cnT + NB_D * NB_K;           // 4*16*64
  unsigned short* at  = wt + 4 * 16 * NB_D;          // 4*16*256
  float* c2  = (float*)(at + 4 * 16 * NB_K);
  float* w2s = c2 + NB_K;

  int tid = threadIdx.x, lane = tid & 31, wave = tid >> 5;
  int ln = lane & 15, hs = lane >> 4;

  for (int i = tid; i < NB_K * NB_D; i += blockDim.x) {
    cl[i] = f2bf(AC[i]);
    int k = i >> 6, d = i & 63;
    cnT[d * NB_K + k] = f2bf(cn[i]);
  }
  for (int k = tid; k < NB_K; k += blockDim.x) {
    float s = 0.f;
    for (int d = 0; d < NB_D; ++d) { float v = AC[k * NB_D + d]; s += v * v; }
    c2[k] = s;
  }
  __syncthreads();

  unsigned short* wtw = wt + wave * 16 * NB_D;
  unsigned short* atw = at + wave * 16 * NB_K;
  float* w2w = w2s + wave * 16;

  for (int tl = 0; tl < 2; ++tl) {
    int rowbase = blockIdx.x * 128 + tl * 64 + wave * 16;

    if (lane < 16) w2w[lane] = 0.f;
    for (int i = 0; i < 32; ++i) {
      int idx = i * 32 + lane;
      float v = W[(size_t)rowbase * NB_D + idx];
      wtw[idx] = f2bf(v);
      atomicAdd(&w2w[idx >> 6], v * v);
    }
    __builtin_prefetch(W + (size_t)((rowbase + 64) % NB_N) * NB_D, 0, 3);

    v8f zf = {0.f, 0.f, 0.f, 0.f, 0.f, 0.f, 0.f, 0.f};
    v8f acc[16];
#pragma unroll
    for (int t = 0; t < 16; ++t) acc[t] = zf;

#pragma unroll
    for (int c = 0; c < 2; ++c) {
      v16bf a = ldfrag(wtw + ln * NB_D + c * 32, hs);
#pragma unroll
      for (int t = 0; t < 16; ++t) {
        v16bf b = ldfrag(cl + (t * 16 + ln) * NB_D + c * 32, hs);
        acc[t] = wmma_bf16(a, b, acc[t]);
      }
    }

    float w2v[8];
#pragma unroll
    for (int r = 0; r < 8; ++r) w2v[r] = w2w[r + 8 * hs];
#pragma unroll
    for (int t = 0; t < 16; ++t) {
      float c2t = c2[t * 16 + ln];
#pragma unroll
      for (int r = 0; r < 8; ++r) {
        float d2 = fmaxf(w2v[r] + c2t - 2.0f * acc[t][r], 0.f);
        acc[t][r] = -sqrtf(d2);
      }
    }

    float mx[8], sm[8];
#pragma unroll
    for (int r = 0; r < 8; ++r) mx[r] = -3.402823466e38f;
#pragma unroll
    for (int t = 0; t < 16; ++t)
#pragma unroll
      for (int r = 0; r < 8; ++r) mx[r] = fmaxf(mx[r], acc[t][r]);
#pragma unroll
    for (int off = 1; off < 16; off <<= 1)
#pragma unroll
      for (int r = 0; r < 8; ++r) mx[r] = fmaxf(mx[r], __shfl_xor(mx[r], off, 16));
#pragma unroll
    for (int r = 0; r < 8; ++r) sm[r] = 0.f;
#pragma unroll
    for (int t = 0; t < 16; ++t)
#pragma unroll
      for (int r = 0; r < 8; ++r) {
        float e = __expf(acc[t][r] - mx[r]);
        acc[t][r] = e;
        sm[r] += e;
      }
#pragma unroll
    for (int off = 1; off < 16; off <<= 1)
#pragma unroll
      for (int r = 0; r < 8; ++r) sm[r] += __shfl_xor(sm[r], off, 16);
#pragma unroll
    for (int r = 0; r < 8; ++r) sm[r] = 1.0f / sm[r];

    // write final attn (f32) and stage bf16 row-major copy for GEMM2 A operand
#pragma unroll
    for (int t = 0; t < 16; ++t)
#pragma unroll
      for (int r = 0; r < 8; ++r) {
        float a = acc[t][r] * sm[r];
        int M = r + 8 * hs, Ncol = t * 16 + ln;
        attn_out[(size_t)(rowbase + M) * NB_K + Ncol] = a;
        atw[M * NB_K + Ncol] = f2bf(a);
      }
    __syncthreads();   // uniform: all waves run the same tl count

    // GEMM2: compressed(16x64) = attn(16x256) @ cn(256x64)
    v8f c2acc[4];
#pragma unroll
    for (int dt = 0; dt < 4; ++dt) c2acc[dt] = zf;
#pragma unroll
    for (int ch = 0; ch < 8; ++ch) {
      v16bf a = ldfrag(atw + ln * NB_K + ch * 32, hs);
#pragma unroll
      for (int dt = 0; dt < 4; ++dt) {
        v16bf b = ldfrag(cnT + (dt * 16 + ln) * NB_K + ch * 32, hs);
        c2acc[dt] = wmma_bf16(a, b, c2acc[dt]);
      }
    }
#pragma unroll
    for (int dt = 0; dt < 4; ++dt)
#pragma unroll
      for (int r = 0; r < 8; ++r)
        comp[(size_t)(rowbase + r + 8 * hs) * NB_D + dt * 16 + ln] = c2acc[dt][r];
  }
}

// ---------------------------------------------------------------------------
extern "C" void kernel_launch(void* const* d_in, const int* in_sizes, int n_in,
                              void* d_out, int out_size, void* d_ws, size_t ws_size,
                              hipStream_t stream) {
  const float* W = (const float*)d_in[0];
  const int* idx = (const int*)d_in[1];

  float* out  = (float*)d_out;
  float* comp = out;                                  // [N, D]
  float* cn   = out + (size_t)NB_N * NB_D;            // [K, D]
  float* attn = cn + NB_K * NB_D;                     // [N, K] (also iter scratch)

  float* C  = (float*)d_ws;                           // [K, D]
  float* AC = C + NB_K * NB_D;                        // [K, D]
  float* WS = AC + NB_K * NB_D;                       // [K, D] wsum/new
  float* CS = WS + NB_K * NB_D;                       // [K]
  int* done = (int*)(CS + NB_K);

  const size_t smem1 = (size_t)NB_K * NB_D * 4 +      // f32 TDM staging
                       (size_t)(NB_K * NB_D + 4 * 16 * NB_D) * 2 +
                       (NB_K + 64 + NB_K) * 4;
  const size_t smem2 = (size_t)(NB_K * 32 + NB_D * 32) * 2;
  const size_t smem5 =
      (size_t)(NB_K * NB_D + NB_D * NB_K + 4 * 16 * NB_D + 4 * 16 * NB_K) * 2 +
      (NB_K + 64) * 4;

  k_init<<<NB_K, NB_D, 0, stream>>>(W, idx, C, AC, done);
  for (int it = 0; it < DKM_ITERS; ++it) {
    k_zero<<<64, 256, 0, stream>>>(WS, CS);
    k_attn_iter<<<NB_N / 128, 128, smem1, stream>>>(W, C, attn, CS, done);
    k_wsum_iter<<<64, 128, smem2, stream>>>(attn, W, WS, done);
    k_update<<<1, 256, 0, stream>>>(C, AC, WS, CS, done);
  }
  k_norm<<<1, NB_K, 0, stream>>>(C, cn);
  k_final<<<NB_N / 128, 128, smem5, stream>>>(W, AC, cn, attn, comp);
}
